// DsAttention_60129542144140
// MI455X (gfx1250) — matrix operand order repaired and checked by reading, not compiled, not run on hardware
//
#include <hip/hip_runtime.h>
#include <hip/hip_bf16.h>

typedef __attribute__((ext_vector_type(16))) _Float16 v16h;
typedef __attribute__((ext_vector_type(8)))  float    v8f;

#define DIM 96
#define NH  6
#define HD  16
#define B4  4

// ---------------------------------------------------------------------------
// generic helpers
__global__ void fill_zero_kernel(unsigned int* __restrict__ p, int n32) {
    int i = blockIdx.x * blockDim.x + threadIdx.x;
    if (i < n32) p[i] = 0u;
}

__global__ void cvt_f32_f16_kernel(const float* __restrict__ s, _Float16* __restrict__ d, int n) {
    int i = blockIdx.x * blockDim.x + threadIdx.x;
    if (i < n) d[i] = (_Float16)s[i];
}

// ---------------------------------------------------------------------------
// x even channels (NCHW f32) -> zero-padded NHWC f16 (B,66,66,96)
__global__ void stage_x1_kernel(const float* __restrict__ x, _Float16* __restrict__ x1p) {
    int idx = blockIdx.x * blockDim.x + threadIdx.x;       // c fastest -> coalesced writes
    if (idx >= B4 * 64 * 64 * DIM) return;
    int c  = idx % DIM;
    int t  = idx / DIM;
    int gx = t & 63;
    int gy = (t >> 6) & 63;
    int b  = t >> 12;
    float v = x[((size_t)(b * 2 * DIM + 2 * c)) * 4096 + gy * 64 + gx];
    x1p[(((size_t)(b * 66 + gy + 1)) * 66 + gx + 1) * DIM + c] = (_Float16)v;
}

// odd channels + 2x2 maxpool -> t2h (B,1024,96) f16, c fastest (coalesced writes)
__global__ void pool_kernel(const float* __restrict__ x, _Float16* __restrict__ t2h) {
    int idx = blockIdx.x * blockDim.x + threadIdx.x;
    if (idx >= B4 * 32 * 32 * DIM) return;
    int c  = idx % DIM;
    int t  = idx / DIM;
    int px = t & 31;
    int py = (t >> 5) & 31;
    int b  = t >> 10;
    const float* xb = x + ((size_t)(b * 2 * DIM + 2 * c + 1)) * 4096;
    int gy = py * 2, gx = px * 2;
    float m0 = fmaxf(xb[gy * 64 + gx],       xb[gy * 64 + gx + 1]);
    float m1 = fmaxf(xb[(gy + 1) * 64 + gx], xb[(gy + 1) * 64 + gx + 1]);
    t2h[((size_t)b * 1024 + py * 32 + px) * DIM + c] = (_Float16)fmaxf(m0, m1);
}

// conv_w (oc,ic,ky,kx) f32 -> cwT (ky*3+kx, oc, ic) f16   (B-operand friendly)
__global__ void stage_cw_kernel(const float* __restrict__ cw, _Float16* __restrict__ cwT) {
    int idx = blockIdx.x * blockDim.x + threadIdx.x;       // ic fastest
    if (idx >= 9 * DIM * DIM) return;
    int ic = idx % DIM;
    int oc = (idx / DIM) % DIM;
    int k9 = idx / (DIM * DIM);
    cwT[((size_t)k9 * DIM + oc) * DIM + ic] = (_Float16)cw[((size_t)(oc * DIM + ic)) * 9 + k9];
}

// ---------------------------------------------------------------------------
// 3x3 SAME conv as 9-shift GEMM, fully unrolled 27 WMMAs, branch-free loads.
// 4 waves / block; wave = 16 pixels (one row segment) x 16 out-channels.
__global__ __launch_bounds__(128) void conv_kernel(const _Float16* __restrict__ x1p,
                                                   const _Float16* __restrict__ cwT,
                                                   const float* __restrict__ cb,
                                                   _Float16* __restrict__ t1h) {
    int w    = threadIdx.x >> 5;
    int lane = threadIdx.x & 31;
    int ln   = lane & 15;
    bool hiw = lane >= 16;
    int mt = blockIdx.x * 4 + w;          // 0..255
    int nt = blockIdx.y;
    int b  = blockIdx.z;
    int y  = mt >> 2;
    int x0 = (mt & 3) << 4;
    int oc0 = nt * 16;
    int co  = hiw ? 8 : 0;

    v8f c = {};
#pragma unroll
    for (int ky = 0; ky < 3; ++ky) {
#pragma unroll
        for (int kx = 0; kx < 3; ++kx) {
            const _Float16* arow = x1p + (((size_t)(b * 66 + y + ky)) * 66 + x0 + ln + kx) * DIM;
            const _Float16* wrow = cwT + ((size_t)(ky * 3 + kx) * DIM + oc0 + ln) * DIM;
#pragma unroll
            for (int kc = 0; kc < 3; ++kc) {
                int c0 = kc * 32;
                v16h a;
#pragma unroll
                for (int j = 0; j < 8; ++j) {
                    a[j]     = arow[c0 + co + j];
                    a[j + 8] = arow[c0 + co + 16 + j];
                }
                v16h bm;
#pragma unroll
                for (int j = 0; j < 16; ++j)
                    bm[j] = wrow[c0 + (hiw ? 8 : 0) + (j < 8 ? j : j + 8)];
                c = __builtin_amdgcn_wmma_f32_16x16x32_f16(false, a, false, bm,
                                                           (short)0, c, false, false);
            }
        }
    }
    float bv = cb[oc0 + ln];
    int l0 = y * 64 + x0;
#pragma unroll
    for (int r = 0; r < 8; ++r) {
        int m = r + (hiw ? 8 : 0);
        t1h[(size_t)b * 4096 * DIM + (size_t)(l0 + m) * DIM + oc0 + ln] = (_Float16)(c[r] + bv);
    }
}

// ---------------------------------------------------------------------------
// Linear projection, f16 weights; optional transposed store (B,NH,HD,L) for V.
__global__ __launch_bounds__(128) void proj_kernel(const _Float16* __restrict__ tin,
                                                   const _Float16* __restrict__ Wh,
                                                   const float* __restrict__ bias,
                                                   _Float16* __restrict__ tout,
                                                   int L, int vt) {
    int w    = threadIdx.x >> 5;
    int lane = threadIdx.x & 31;
    int ln   = lane & 15;
    bool hiw = lane >= 16;
    int mt = blockIdx.x * 4 + w;
    int nt = blockIdx.y;
    int b  = blockIdx.z;
    int l0 = mt * 16, d0 = nt * 16;
    int co = hiw ? 8 : 0;
    const _Float16* tb = tin + (size_t)b * L * DIM + (size_t)(l0 + ln) * DIM;
    const _Float16* wr = Wh + (size_t)(d0 + ln) * DIM + (hiw ? 16 : 0);

    v8f c = {};
#pragma unroll
    for (int kc = 0; kc < 3; ++kc) {
        int c0 = kc * 32;
        v16h a;
#pragma unroll
        for (int j = 0; j < 8; ++j) {
            a[j]     = tb[c0 + 2 * co + j];
            a[j + 8] = tb[c0 + 2 * co + 8 + j];
        }
        v16h bm;
#pragma unroll
        for (int j = 0; j < 16; ++j)
            bm[j] = wr[c0 + j];
        c = __builtin_amdgcn_wmma_f32_16x16x32_f16(false, a, false, bm,
                                                   (short)0, c, false, false);
    }
    float bv = bias[d0 + ln];
    if (!vt) {
#pragma unroll
        for (int r = 0; r < 8; ++r) {
            int m = r + (hiw ? 8 : 0);
            tout[(size_t)b * L * DIM + (size_t)(l0 + m) * DIM + d0 + ln] = (_Float16)(c[r] + bv);
        }
    } else {
        // transposed: Vt[b][head=nt][d=ln][l]  (contiguous in l per lane)
        _Float16* vp = tout + ((size_t)(b * NH + nt) * HD + ln) * L + l0 + (hiw ? 8 : 0);
#pragma unroll
        for (int r = 0; r < 8; ++r)
            vp[r] = (_Float16)(c[r] + bv);
    }
}

// ---------------------------------------------------------------------------
// Flash cross-attention, QB Q-tiles (QB*16 rows) per wave so K/V chunk loads
// are reused QB times. 4 waves / block. Per 32-key chunk: 2*QB score WMMAs +
// QB PV WMMAs off one bk0/bk1/bv fetch. Full-wave softmax: lane ln & ln+16
// split the 32 columns of row ln, combined via shfl_xor(16).
template <int QB>
__global__ __launch_bounds__(128) void attn_kernel(const _Float16* __restrict__ Qh,
                                                   const _Float16* __restrict__ Kh,
                                                   const _Float16* __restrict__ Vt,
                                                   int Lq, int Lk,
                                                   float* __restrict__ outp, int upsample) {
    __shared__ alignas(64) float    s_s[4][16][32];
    __shared__ alignas(64) _Float16 s_p[4][16][32];
    __shared__ float s_scale[4][16];
    __shared__ float s_lsum[4][16];

    int w    = threadIdx.x >> 5;
    int lane = threadIdx.x & 31;
    int ln   = lane & 15;
    bool hiw = lane >= 16;
    int ro   = hiw ? 8 : 0;            // accumulator row offset
    int cb   = hiw ? 16 : 0;           // softmax column base
    int qg   = blockIdx.x * 4 + w;     // group of QB tiles
    int h    = blockIdx.y;
    int b    = blockIdx.z;
    const _Float16* Qb  = Qh + (size_t)b * Lq * DIM + h * HD;
    const _Float16* Kb  = Kh + (size_t)b * Lk * DIM + h * HD;
    const _Float16* Vtb = Vt + ((size_t)(b * NH + h) * HD + ln) * Lk;   // lane's V row (d=ln)

    // Q A-operands: halves 0..7 -> gemm-K 0..15 (head dims); halves 8..15 = 0
    v16h aq[QB];
    v8f  acc[QB] = {};
    float mrow[QB], lrow[QB];
#pragma unroll
    for (int qi = 0; qi < QB; ++qi) {
        const _Float16* qrow = Qb + (size_t)((qg * QB + qi) * 16 + ln) * DIM + (hiw ? 8 : 0);
#pragma unroll
        for (int j = 0; j < 8; ++j) {
            aq[qi][j]     = qrow[j];
            aq[qi][j + 8] = (_Float16)0.0f;
        }
        mrow[qi] = -1e30f;
        lrow[qi] = 0.0f;
    }

    for (int kk = 0; kk < Lk; kk += 32) {
        if (kk + 32 < Lk) {
            __builtin_prefetch(Kb + (size_t)(kk + 32 + ln) * DIM, 0, 1);
            __builtin_prefetch(Vtb + kk + 32, 0, 1);
        }
        // one K/V chunk fetch, reused for all QB tiles
        v16h bk0, bk1, bv;
        {
            const _Float16* k0 = Kb + (size_t)(kk + ln) * DIM + (hiw ? 8 : 0);
            const _Float16* k1 = Kb + (size_t)(kk + 16 + ln) * DIM + (hiw ? 8 : 0);
            const _Float16* vr = Vtb + kk + cb;
#pragma unroll
            for (int j = 0; j < 16; ++j) {
                bk0[j] = k0[j];
                bk1[j] = k1[j];
                bv[j]  = vr[j];
            }
        }
#pragma unroll
        for (int qi = 0; qi < QB; ++qi) {
            // scores (B upper-K halves are don't-care: A is zero there)
            v8f s0 = {}, s1 = {};
            s0 = __builtin_amdgcn_wmma_f32_16x16x32_f16(false, aq[qi], false, bk0,
                                                        (short)0, s0, false, false);
            s1 = __builtin_amdgcn_wmma_f32_16x16x32_f16(false, aq[qi], false, bk1,
                                                        (short)0, s1, false, false);
#pragma unroll
            for (int r = 0; r < 8; ++r) {
                s_s[w][r + ro][ln]      = s0[r] * 0.25f;
                s_s[w][r + ro][16 + ln] = s1[r] * 0.25f;
            }
            __builtin_amdgcn_wave_barrier();

            // online softmax: lane handles row ln, columns [cb, cb+16)
            float mx = mrow[qi];
#pragma unroll
            for (int jc = 0; jc < 16; ++jc) mx = fmaxf(mx, s_s[w][ln][cb + jc]);
            mx = fmaxf(mx, __shfl_xor(mx, 16, 32));
            float sc = __expf(mrow[qi] - mx);
            float sum = 0.0f;
#pragma unroll
            for (int jc = 0; jc < 16; ++jc) {
                float p = __expf(s_s[w][ln][cb + jc] - mx);
                sum += p;
                s_p[w][ln][cb + jc] = (_Float16)p;
            }
            sum += __shfl_xor(sum, 16, 32);
            mrow[qi] = mx;
            lrow[qi] = lrow[qi] * sc + sum;
            if (lane < 16) s_scale[w][lane] = sc;
            __builtin_amdgcn_wave_barrier();

            // rescale accumulator, then acc += P * V
#pragma unroll
            for (int r = 0; r < 8; ++r)
                acc[qi][r] *= s_scale[w][r + ro];

            v16h ap;
            {
                const _Float16* pr = &s_p[w][ln][hiw ? 16 : 0];
#pragma unroll
                for (int j = 0; j < 8; ++j) {
                    ap[j]     = pr[j];
                    ap[j + 8] = pr[8 + j];
                }
            }
            acc[qi] = __builtin_amdgcn_wmma_f32_16x16x32_f16(false, ap, false, bv,
                                                             (short)0, acc[qi], false, false);
            __builtin_amdgcn_wave_barrier();
        }
    }

    // normalize + NCHW store (o2: nearest-neighbor 2x upsample)
#pragma unroll
    for (int qi = 0; qi < QB; ++qi) {
        if (lane < 16) s_lsum[w][lane] = lrow[qi];
        __builtin_amdgcn_wave_barrier();
#pragma unroll
        for (int r = 0; r < 8; ++r) {
            int m = r + ro;
            float o = acc[qi][r] / s_lsum[w][m];
            int l = (qg * QB + qi) * 16 + m;
            int cdim = h * HD + ln;
            if (!upsample) {
                outp[((size_t)b * 2 * DIM + cdim) * 4096 + l] = o;
            } else {
                int py = l >> 5, px = l & 31;
                float* p0 = outp + ((size_t)b * 2 * DIM + DIM + cdim) * 4096 + (py * 2) * 64 + px * 2;
                p0[0] = o; p0[1] = o; p0[64] = o; p0[65] = o;
            }
        }
        __builtin_amdgcn_wave_barrier();
    }
}

// ---------------------------------------------------------------------------
extern "C" void kernel_launch(void* const* d_in, const int* in_sizes, int n_in,
                              void* d_out, int out_size, void* d_ws, size_t ws_size,
                              hipStream_t stream) {
    (void)in_sizes; (void)n_in; (void)out_size; (void)ws_size;
    const float* x      = (const float*)d_in[0];
    const float* conv_w = (const float*)d_in[1];
    const float* conv_b = (const float*)d_in[2];
    const float* pw[6]  = { (const float*)d_in[3], (const float*)d_in[5], (const float*)d_in[7],
                            (const float*)d_in[9], (const float*)d_in[11], (const float*)d_in[13] };
    const float* pb[6]  = { (const float*)d_in[4], (const float*)d_in[6], (const float*)d_in[8],
                            (const float*)d_in[10], (const float*)d_in[12], (const float*)d_in[14] };
    float* out = (float*)d_out;

    const size_t L1 = 4096, L2 = 1024;
    const size_t szL1 = B4 * L1 * DIM;           // f16 elements
    const size_t szL2 = B4 * L2 * DIM;
    const size_t szPad = (size_t)B4 * 66 * 66 * DIM;

    char* ws = (char*)d_ws;
    _Float16* x1p = (_Float16*)ws;  ws += szPad * 2;         // padded NHWC conv input
    _Float16* cwT = (_Float16*)ws;  ws += (size_t)9 * DIM * DIM * 2;
    _Float16* Wh6 = (_Float16*)ws;  ws += (size_t)6 * DIM * DIM * 2;
    _Float16* t1h = (_Float16*)ws;  ws += szL1 * 2;
    _Float16* t2h = (_Float16*)ws;  ws += szL2 * 2;
    _Float16* LQ  = (_Float16*)ws;  ws += szL1 * 2;
    _Float16* LK  = (_Float16*)ws;  ws += szL1 * 2;
    _Float16* LVt = (_Float16*)ws;  ws += szL1 * 2;          // transposed (B,NH,HD,L1)
    _Float16* GQ  = (_Float16*)ws;  ws += szL2 * 2;
    _Float16* GK  = (_Float16*)ws;  ws += szL2 * 2;
    _Float16* GVt = (_Float16*)ws;  ws += szL2 * 2;          // transposed (B,NH,HD,L2)

    // 1) staging: zero pad buffer, convert x1, pool x2, transform weights
    int n32 = (int)(szPad * 2 / 4);
    fill_zero_kernel<<<(n32 + 255) / 256, 256, 0, stream>>>((unsigned int*)x1p, n32);
    int nx1 = B4 * 64 * 64 * DIM;
    stage_x1_kernel<<<(nx1 + 255) / 256, 256, 0, stream>>>(x, x1p);
    int npool = B4 * 32 * 32 * DIM;
    pool_kernel<<<(npool + 255) / 256, 256, 0, stream>>>(x, t2h);
    int ncw = 9 * DIM * DIM;
    stage_cw_kernel<<<(ncw + 255) / 256, 256, 0, stream>>>(conv_w, cwT);
    for (int i = 0; i < 6; ++i)
        cvt_f32_f16_kernel<<<(DIM * DIM + 255) / 256, 256, 0, stream>>>(
            pw[i], Wh6 + (size_t)i * DIM * DIM, DIM * DIM);

    // 2) conv -> t1h (27 straight-line WMMAs per wave)
    conv_kernel<<<dim3(64, NH, B4), 128, 0, stream>>>(x1p, cwT, conv_b, t1h);

    // 3) projections (WMMA, f16 weights); V stored transposed
    _Float16* Wq = Wh6 + 0 * DIM * DIM, *Wk = Wh6 + 1 * DIM * DIM, *Wv = Wh6 + 2 * DIM * DIM;
    _Float16* Gq = Wh6 + 3 * DIM * DIM, *Gk = Wh6 + 4 * DIM * DIM, *Gv = Wh6 + 5 * DIM * DIM;
    proj_kernel<<<dim3(64, NH, B4), 128, 0, stream>>>(t1h, Wq, pb[0], LQ,  (int)L1, 0);
    proj_kernel<<<dim3(64, NH, B4), 128, 0, stream>>>(t1h, Wk, pb[1], LK,  (int)L1, 0);
    proj_kernel<<<dim3(64, NH, B4), 128, 0, stream>>>(t1h, Wv, pb[2], LVt, (int)L1, 1);
    proj_kernel<<<dim3(16, NH, B4), 128, 0, stream>>>(t2h, Gq, pb[3], GQ,  (int)L2, 0);
    proj_kernel<<<dim3(16, NH, B4), 128, 0, stream>>>(t2h, Gk, pb[4], GK,  (int)L2, 0);
    proj_kernel<<<dim3(16, NH, B4), 128, 0, stream>>>(t2h, Gv, pb[5], GVt, (int)L2, 1);

    // 4) o1 = attn(LQ, GK, GV): 4096q x 1024k, QB=4 -> 16 blocks
    attn_kernel<4><<<dim3(16, NH, B4), 128, 0, stream>>>(LQ, GK, GVt, (int)L1, (int)L2, out, 0);
    // 5) o2 = attn(GQ, LK, LV): 1024q x 4096k, QB=2 -> 8 blocks, 2x2 upsample
    attn_kernel<2><<<dim3(8, NH, B4), 128, 0, stream>>>(GQ, LK, LVt, (int)L2, (int)L1, out, 1);
}
